// ocvqa_32246614458834
// MI455X (gfx1250) — matrix operand order repaired and checked by reading, not compiled, not run on hardware
//
#include <hip/hip_runtime.h>

#define B_  256
#define S_  32
#define D_  64
#define HQ_ 128
#define RD_ 256
#define AV_ 1000

typedef __bf16 bf16_t;
typedef __attribute__((ext_vector_type(16))) __bf16 v16bf;
typedef __attribute__((ext_vector_type(8)))  __bf16 v8bf;
typedef __attribute__((ext_vector_type(4)))  __bf16 v4bf;
typedef __attribute__((ext_vector_type(8)))  float  v8f;
typedef __attribute__((ext_vector_type(4)))  int    v4i;

#define AS1 __attribute__((address_space(1)))
#define AS3 __attribute__((address_space(3)))

#define LDH 264   // padded row stride (elements) for activation tile
#define LDW 264   // padded row stride (elements) for weight tile
#define SMEM_MAIN ((128*LDH + 256*LDW) * (int)sizeof(bf16_t))

#if defined(__gfx1250__) && __has_builtin(__builtin_amdgcn_global_load_async_to_lds_b128)
#define RN_HAVE_ASYNC_LDS 1
#else
#define RN_HAVE_ASYNC_LDS 0
#endif

__device__ __forceinline__ void rn_wait_asynccnt0() {
#if __has_builtin(__builtin_amdgcn_s_wait_asynccnt)
    __builtin_amdgcn_s_wait_asynccnt(0);
#else
    asm volatile("s_wait_asynccnt 0" ::: "memory");
#endif
}

__device__ __forceinline__ v16bf rn_load_bfrag(const bf16_t* p) {
    v8bf lo = *(const v8bf*)(p);
    v8bf hi = *(const v8bf*)(p + 16);
    return __builtin_shufflevector(lo, hi,
            0,1,2,3,4,5,6,7,8,9,10,11,12,13,14,15);
}

// Force issue order  DS,DS, (DS,DS,WMMA)x7, WMMA  inside the nt-loop body so
// the B-fragment loads for kstep ks+1 are in flight while WMMA(ks) executes.
__device__ __forceinline__ void rn_pipeline_hint() {
#if __has_builtin(__builtin_amdgcn_sched_group_barrier)
    __builtin_amdgcn_sched_group_barrier(0x100, 2, 0);   // 2 x DS read
    #pragma unroll
    for (int g = 0; g < 7; ++g) {
        __builtin_amdgcn_sched_group_barrier(0x100, 2, 0); // 2 x DS read
        __builtin_amdgcn_sched_group_barrier(0x008, 1, 0); // 1 x WMMA
    }
    __builtin_amdgcn_sched_group_barrier(0x008, 1, 0);     // final WMMA
#endif
}

// ---------------------------------------------------------------------------
// Kernel 1: transpose+convert W_r2/3/4 (f32 [K][N]) -> bf16 [N][K]
// ---------------------------------------------------------------------------
__global__ void rn_prep_weights(const float* __restrict__ W2,
                                const float* __restrict__ W3,
                                const float* __restrict__ W4,
                                bf16_t* __restrict__ WT2,
                                bf16_t* __restrict__ WT3,
                                bf16_t* __restrict__ WT4) {
    const int k = blockIdx.x;        // 0..255
    const int l = blockIdx.y;        // 0..2
    const float* W  = (l == 0) ? W2  : (l == 1) ? W3  : W4;
    bf16_t*      WT = (l == 0) ? WT2 : (l == 1) ? WT3 : WT4;
    const int n = threadIdx.x;       // 0..255
    WT[n * RD_ + k] = (bf16_t)W[k * RD_ + n];
}

// ---------------------------------------------------------------------------
// Kernel 2: A1[b,i,n] = obj[b,i,:]@Wa[:,n]
//           A2[b,j,n] = obj[b,j,:]@Wb[:,n] + q[b,:]@Wq[:,n] + b_r1[n]
// also zero-inits relations accumulator
// ---------------------------------------------------------------------------
__global__ __launch_bounds__(256) void rn_phase_a(
    const float* __restrict__ objects, const float* __restrict__ qf,
    const float* __restrict__ W1, const float* __restrict__ b1,
    float* __restrict__ A1, float* __restrict__ A2, float* __restrict__ rel) {
    __shared__ float sObj[S_ * D_];
    __shared__ float sQ[HQ_];
    const int b = blockIdx.x;
    const int t = threadIdx.x;       // output column n
    for (int e = t; e < S_ * D_; e += 256) sObj[e] = objects[b * S_ * D_ + e];
    if (t < HQ_) sQ[t] = qf[b * HQ_ + t];
    rel[b * RD_ + t] = 0.0f;
    __syncthreads();

    float acc[S_];
    // ha = obj @ Wa   (Wa = W_r1 rows [0,64))
    #pragma unroll
    for (int i = 0; i < S_; ++i) acc[i] = 0.0f;
    for (int d = 0; d < D_; ++d) {
        const float w = W1[d * RD_ + t];
        #pragma unroll
        for (int i = 0; i < S_; ++i) acc[i] += sObj[i * D_ + d] * w;
    }
    for (int i = 0; i < S_; ++i) A1[(b * S_ + i) * RD_ + t] = acc[i];

    // hb = obj @ Wb   (Wb = W_r1 rows [64,128))
    #pragma unroll
    for (int i = 0; i < S_; ++i) acc[i] = 0.0f;
    for (int d = 0; d < D_; ++d) {
        const float w = W1[(D_ + d) * RD_ + t];
        #pragma unroll
        for (int i = 0; i < S_; ++i) acc[i] += sObj[i * D_ + d] * w;
    }
    // hq + bias folded into A2
    float hq = b1[t];
    for (int h = 0; h < HQ_; ++h) hq += sQ[h] * W1[(2 * D_ + h) * RD_ + t];
    for (int j = 0; j < S_; ++j) A2[(b * S_ + j) * RD_ + t] = acc[j] + hq;
}

// ---------------------------------------------------------------------------
// Helpers for the main WMMA kernel
// ---------------------------------------------------------------------------
// Stage full 256x256 bf16 weight matrix (global [N][K]) into LDS [N][LDW].
__device__ __forceinline__ void rn_stage_w(bf16_t* sW, const bf16_t* WT, int t) {
    __syncthreads();   // everyone done with previous sW / sH reads
#if RN_HAVE_ASYNC_LDS
    for (int c = t; c < (RD_ * RD_) / 8; c += 256) {
        const int n  = c >> 5;
        const int ko = (c & 31) * 8;
        AS1 v4i* g = (AS1 v4i*)(unsigned long long)(const void*)(WT + n * RD_ + ko);
        AS3 v4i* l = (AS3 v4i*)(unsigned int)(unsigned long long)(void*)(sW + n * LDW + ko);
        __builtin_amdgcn_global_load_async_to_lds_b128(g, l, 0, 0);
    }
    rn_wait_asynccnt0();
#else
    __builtin_prefetch(WT, 0, 1);
    for (int c = t; c < (RD_ * RD_) / 8; c += 256) {
        const int n  = c >> 5;
        const int ko = (c & 31) * 8;
        *(v8bf*)(sW + n * LDW + ko) = *(const v8bf*)(WT + n * RD_ + ko);
    }
#endif
    __syncthreads();
}

// One dense layer: Hnext = relu(H @ W + bias), in-place on sH.
// FINAL: instead of storing, reduce rows into LDS float accumulator relf[256].
template <bool FINAL>
__device__ __forceinline__ void rn_layer_compute(
    bf16_t* sH, const bf16_t* sW, const float* __restrict__ bias,
    float* relf, int wave, int l16, int half) {
    // A fragments: this wave's private 16 rows, all K (8 ksteps)
    const bf16_t* rowp = sH + (wave * 16 + l16) * LDH;
    v16bf a[8];
    #pragma unroll
    for (int ks = 0; ks < 8; ++ks) {
        v8bf lo = *(const v8bf*)(rowp + ks * 32 + half * 8);
        v8bf hi = *(const v8bf*)(rowp + ks * 32 + 16 + half * 8);
        a[ks] = __builtin_shufflevector(lo, hi,
                 0,1,2,3,4,5,6,7,8,9,10,11,12,13,14,15);
    }
    if (FINAL) {
        // A-frags of all waves are in registers now; reuse head of sH as f32
        // partial-sum buffer. Barrier makes all pending reads of sH complete.
        __syncthreads();
        relf[threadIdx.x] = 0.0f;
        __syncthreads();
    }

    #pragma unroll 2
    for (int nt = 0; nt < 16; ++nt) {
        const float bv = bias[nt * 16 + l16];   // issued early, used late
        const bf16_t* wp0 = sW + (nt * 16 + l16) * LDW + half * 8;
        // Rotating 2-deep B pipeline: B(ks+1) is loaded (live in a second
        // buffer) before wmma(ks) consumes B(ks) -> non-blocking dscnt waits.
        v16bf bcur = rn_load_bfrag(wp0);
        v8f acc = {0.f,0.f,0.f,0.f,0.f,0.f,0.f,0.f};
        #pragma unroll
        for (int ks = 0; ks < 8; ++ks) {
            v16bf bnext = bcur;
            if (ks < 7) bnext = rn_load_bfrag(wp0 + (ks + 1) * 32);
            acc = __builtin_amdgcn_wmma_f32_16x16x32_bf16(
                    false, a[ks], false, bcur, (short)0, acc, false, false);
            bcur = bnext;
        }
        rn_pipeline_hint();
        if (!FINAL) {
            // D layout: VGPR v -> row wave*16 + half*8 + v, col nt*16 + l16
            #pragma unroll
            for (int v = 0; v < 8; ++v) {
                const float x = fmaxf(acc[v] + bv, 0.0f);
                sH[(wave * 16 + half * 8 + v) * LDH + nt * 16 + l16] = (bf16_t)x;
            }
        } else {
            float s = 0.0f;
            #pragma unroll
            for (int v = 0; v < 8; ++v) s += fmaxf(acc[v] + bv, 0.0f);
            atomicAdd(&relf[nt * 16 + l16], s);   // ds_add_f32
        }
    }
}

// ---------------------------------------------------------------------------
// Kernel 3: fused 3-layer pairwise MLP + row-sum, bf16 WMMA
//   grid = (8 pair-blocks of 128, 256 batches), 256 thr = 8 waves
// ---------------------------------------------------------------------------
__global__ __launch_bounds__(256) void rn_main(
    const float* __restrict__ A1, const float* __restrict__ A2,
    const bf16_t* __restrict__ WT2, const bf16_t* __restrict__ WT3,
    const bf16_t* __restrict__ WT4,
    const float* __restrict__ b2, const float* __restrict__ b3,
    const float* __restrict__ b4,
    float* __restrict__ rel) {
    extern __shared__ char smem[];
    bf16_t* sH = (bf16_t*)smem;          // 128 x LDH activations (bf16)
    bf16_t* sW = sH + 128 * LDH;         // 256 x LDW weights (bf16, [N][K])
    float*  relf = (float*)smem;         // reused by final layer reduction

    const int b    = blockIdx.y;
    const int pblk = blockIdx.x;         // 0..7 -> rows pblk*128 .. +127
    const int t    = threadIdx.x;
    const int wave = t >> 5;
    const int lane = t & 31;
    const int half = lane >> 4;
    const int l16  = lane & 15;

    // ---- build initial H = relu(A1[i] + A2[j]), 4 cols per thread (float4)
    const float* A1b = A1 + (size_t)b * S_ * RD_;
    const float* A2b = A2 + (size_t)b * S_ * RD_;
    const int c4 = (t & 63) * 4;         // column group
    const int tr = t >> 6;               // row within group of 4
    for (int it = 0; it < 32; ++it) {
        const int r = it * 4 + tr;
        const int p = pblk * 128 + r;
        const int i = p >> 5, j = p & 31;
        const float4 va = *(const float4*)(A1b + i * RD_ + c4);
        const float4 vb = *(const float4*)(A2b + j * RD_ + c4);
        v4bf hv;
        hv[0] = (bf16_t)fmaxf(va.x + vb.x, 0.0f);
        hv[1] = (bf16_t)fmaxf(va.y + vb.y, 0.0f);
        hv[2] = (bf16_t)fmaxf(va.z + vb.z, 0.0f);
        hv[3] = (bf16_t)fmaxf(va.w + vb.w, 0.0f);
        *(v4bf*)(sH + r * LDH + c4) = hv;
    }

    // ---- layer 1 (W_r2) and layer 2 (W_r3): store back to sH
    rn_stage_w(sW, WT2, t);
    rn_layer_compute<false>(sH, sW, b2, relf, wave, l16, half);
    rn_stage_w(sW, WT3, t);
    rn_layer_compute<false>(sH, sW, b3, relf, wave, l16, half);
    // ---- layer 3 (W_r4): reduce rows into LDS, then one global atomic/col
    rn_stage_w(sW, WT4, t);
    rn_layer_compute<true>(sH, sW, b4, relf, wave, l16, half);
    __syncthreads();
    atomicAdd(&rel[b * RD_ + t], relf[t]);
}

// ---------------------------------------------------------------------------
// Kernel 4: head — p1,p2,p3 MLP + softmax, one block per batch row
// ---------------------------------------------------------------------------
__global__ __launch_bounds__(256) void rn_head(
    const float* __restrict__ rel,
    const float* __restrict__ Wp1, const float* __restrict__ bp1,
    const float* __restrict__ Wp2, const float* __restrict__ bp2,
    const float* __restrict__ Wp3, const float* __restrict__ bp3,
    const float* __restrict__ Wout, const float* __restrict__ bout,
    float* __restrict__ out) {
    __shared__ float x[256];
    __shared__ float y[256];
    __shared__ float r3[32];
    __shared__ float red[256];
    const int b = blockIdx.x, t = threadIdx.x;

    x[t] = rel[b * RD_ + t];
    __syncthreads();
    {   float s = bp1[t];
        for (int k = 0; k < 256; ++k) s += x[k] * Wp1[k * 256 + t];
        y[t] = fmaxf(s, 0.0f); }
    __syncthreads();
    x[t] = y[t];
    __syncthreads();
    {   float s = bp2[t];
        for (int k = 0; k < 256; ++k) s += x[k] * Wp2[k * 256 + t];
        y[t] = fmaxf(s, 0.0f); }
    __syncthreads();
    x[t] = y[t];
    __syncthreads();
    if (t < 29) {
        float s = bp3[t];
        for (int k = 0; k < 256; ++k) s += x[k] * Wp3[k * 29 + t];
        r3[t] = fmaxf(s, 0.0f);
    }
    __syncthreads();

    float lg[4];
    #pragma unroll
    for (int u = 0; u < 4; ++u) {
        const int n = t + u * 256;
        float s = -1e30f;
        if (n < AV_) {
            s = bout[n];
            for (int k = 0; k < 29; ++k) s += r3[k] * Wout[k * AV_ + n];
        }
        lg[u] = s;
    }
    float m = fmaxf(fmaxf(lg[0], lg[1]), fmaxf(lg[2], lg[3]));
    red[t] = m; __syncthreads();
    for (int off = 128; off > 0; off >>= 1) {
        if (t < off) red[t] = fmaxf(red[t], red[t + off]);
        __syncthreads();
    }
    m = red[0]; __syncthreads();
    float es[4], sum = 0.0f;
    #pragma unroll
    for (int u = 0; u < 4; ++u) {
        const int n = t + u * 256;
        const float e = (n < AV_) ? __expf(lg[u] - m) : 0.0f;
        es[u] = e; sum += e;
    }
    red[t] = sum; __syncthreads();
    for (int off = 128; off > 0; off >>= 1) {
        if (t < off) red[t] += red[t + off];
        __syncthreads();
    }
    const float inv = 1.0f / red[0];
    #pragma unroll
    for (int u = 0; u < 4; ++u) {
        const int n = t + u * 256;
        if (n < AV_) out[b * AV_ + n] = es[u] * inv;
    }
}

// ---------------------------------------------------------------------------
extern "C" void kernel_launch(void* const* d_in, const int* in_sizes, int n_in,
                              void* d_out, int out_size, void* d_ws, size_t ws_size,
                              hipStream_t stream) {
    const float* objects = (const float*)d_in[0];
    const float* q_feat  = (const float*)d_in[1];
    const float* W_r1 = (const float*)d_in[2];   const float* b_r1 = (const float*)d_in[3];
    const float* W_r2 = (const float*)d_in[4];   const float* b_r2 = (const float*)d_in[5];
    const float* W_r3 = (const float*)d_in[6];   const float* b_r3 = (const float*)d_in[7];
    const float* W_r4 = (const float*)d_in[8];   const float* b_r4 = (const float*)d_in[9];
    const float* W_p1 = (const float*)d_in[10];  const float* b_p1 = (const float*)d_in[11];
    const float* W_p2 = (const float*)d_in[12];  const float* b_p2 = (const float*)d_in[13];
    const float* W_p3 = (const float*)d_in[14];  const float* b_p3 = (const float*)d_in[15];
    const float* W_out= (const float*)d_in[16];  const float* b_out= (const float*)d_in[17];
    float* out = (float*)d_out;

    char* ws = (char*)d_ws;
    size_t off = 0;
    auto alloc = [&](size_t bytes) -> void* {
        void* p = ws + off;
        off += (bytes + 255) & ~(size_t)255;
        return p;
    };
    float*  A1  = (float*)alloc(sizeof(float) * B_ * S_ * RD_);
    float*  A2  = (float*)alloc(sizeof(float) * B_ * S_ * RD_);
    bf16_t* WT2 = (bf16_t*)alloc(sizeof(bf16_t) * RD_ * RD_);
    bf16_t* WT3 = (bf16_t*)alloc(sizeof(bf16_t) * RD_ * RD_);
    bf16_t* WT4 = (bf16_t*)alloc(sizeof(bf16_t) * RD_ * RD_);
    float*  rel = (float*)alloc(sizeof(float) * B_ * RD_);
    (void)ws_size; (void)in_sizes; (void)n_in; (void)out_size;

    (void)hipFuncSetAttribute(reinterpret_cast<const void*>(rn_main),
                              hipFuncAttributeMaxDynamicSharedMemorySize, SMEM_MAIN);

    rn_prep_weights<<<dim3(RD_, 3), 256, 0, stream>>>(W_r2, W_r3, W_r4, WT2, WT3, WT4);
    rn_phase_a<<<B_, 256, 0, stream>>>(objects, q_feat, W_r1, b_r1, A1, A2, rel);
    rn_main<<<dim3(8, B_), 256, SMEM_MAIN, stream>>>(A1, A2, WT2, WT3, WT4,
                                                     b_r2, b_r3, b_r4, rel);
    rn_head<<<B_, 256, 0, stream>>>(rel, W_p1, b_p1, W_p2, b_p2,
                                    W_p3, b_p3, W_out, b_out, out);
}
